// MultiHeadAttention_66048007078147
// MI455X (gfx1250) — compile-verified, matrix operand order checked by reading, not run on hardware
//
#include <hip/hip_runtime.h>
#include <hip/hip_bf16.h>

// ---------------------------------------------------------------------------
// MHA for MI455X (gfx1250): bf16 WMMA (16x16x32) matmuls w/ fp32 accumulate.
// Pipeline: 3x gemm (QKV proj, bf16 head-major out) -> fused scores/softmax/
// attn-write/PV kernel (TDM double-buffered K stream) -> 1x gemm (out proj).
// ---------------------------------------------------------------------------

typedef __bf16 bf16_t;
typedef bf16_t bf16x8  __attribute__((ext_vector_type(8)));
typedef bf16_t bf16x16 __attribute__((ext_vector_type(16)));
typedef float  floatx8 __attribute__((ext_vector_type(8)));
typedef float  v8f     __attribute__((ext_vector_type(8)));
typedef unsigned int uint32x4 __attribute__((ext_vector_type(4)));
typedef int          int32x8  __attribute__((ext_vector_type(8)));
typedef int          int32x4  __attribute__((ext_vector_type(4)));

#define B_  8
#define L_  1024
#define D_  1024
#define H_  16
#define DH_ 64

#if defined(__gfx1250__) && __has_builtin(__builtin_amdgcn_tensor_load_to_lds)
#define HAVE_TDM 1
#else
#define HAVE_TDM 0
#endif

__device__ __forceinline__ v8f wmma_bf16(bf16x16 a, bf16x16 b, v8f c) {
  return __builtin_amdgcn_wmma_f32_16x16x32_bf16(
      false, a, false, b, (short)0, c, false, false);
}

// ISA A/B fragment layout: lanes 0-15 K={0..7,16..23}, lanes 16-31 +8.
__device__ __forceinline__ bf16x16 frag_ld(const bf16_t* base, int ldm,
                                           int row0, int k0, int lane) {
  const int half = (lane >> 4) & 1;
  const bf16_t* p = base + (size_t)(row0 + (lane & 15)) * ldm + k0 + half * 8;
  bf16x8 lo = *reinterpret_cast<const bf16x8*>(p);
  bf16x8 hi = *reinterpret_cast<const bf16x8*>(p + 16);
  return __builtin_shufflevector(lo, hi, 0, 1, 2, 3, 4, 5, 6, 7,
                                 8, 9, 10, 11, 12, 13, 14, 15);
}

__device__ __forceinline__ bf16x16 frag_ld_f32(const float* base, int ldm,
                                               int row0, int k0, int lane) {
  const int half = (lane >> 4) & 1;
  const float* p = base + (size_t)(row0 + (lane & 15)) * ldm + k0 + half * 8;
  floatx8 lo = *reinterpret_cast<const floatx8*>(p);
  floatx8 hi = *reinterpret_cast<const floatx8*>(p + 16);
  bf16x16 f;
#pragma unroll
  for (int i = 0; i < 8; i++) {
    f[i]     = (bf16_t)lo[i];
    f[8 + i] = (bf16_t)hi[i];
  }
  return f;
}

#if HAVE_TDM
// LDS aperture maps generic addr[31:0] to the LDS byte offset (ISA 10.2).
__device__ __forceinline__ unsigned lds_off(const void* p) {
  return (unsigned)(unsigned long long)p;
}

// TDM 2-D bf16 tile load, D# groups per CDNA5 ISA.
// Pads LDS rows: 128B of data (32 dw) + 16B pad (4 dw) -> pitch 72 bf16,
// matching frag_ld's conflict-free leading dim.
__device__ __forceinline__ void tdm_load_2d_bf16(
    unsigned lds_addr, const void* gptr, unsigned tensor_d0, unsigned tensor_d1,
    unsigned tile_d0, unsigned tile_d1, unsigned stride0) {
  unsigned long long ga = (unsigned long long)gptr;
  uint32x4 g0;
  g0[0] = 1u;                                   // count=1, no gather
  g0[1] = lds_addr;                             // lds_addr [63:32]
  g0[2] = (unsigned)(ga & 0xFFFFFFFFu);         // global_addr [95:64]
  g0[3] = (unsigned)((ga >> 32) & 0x01FFFFFFu)  // global_addr [120:96]
          | (2u << 30);                         // type=2 (image)
  int32x8 g1;
  g1[0] = (int)((1u << 16)                      // data_size=1 (2 bytes)
                | (1u << 20)                    // pad_enable
                | (4u << 22)                    // pad_interval: 32 dwords
                | (3u << 25));                  // pad_amount: 4 dwords
  g1[1] = (int)((tensor_d0 & 0xFFFFu) << 16);   // tensor_dim0 lo @ bit 48
  g1[2] = (int)(((tensor_d0 >> 16) & 0xFFFFu) | ((tensor_d1 & 0xFFFFu) << 16));
  g1[3] = (int)(((tensor_d1 >> 16) & 0xFFFFu) | ((tile_d0 & 0xFFFFu) << 16));
  g1[4] = (int)(tile_d1 & 0xFFFFu);             // tile_dim1; tile_dim2=0
  g1[5] = (int)stride0;                         // tensor_dim0_stride lo
  g1[6] = 0;
  g1[7] = 0;
  int32x4 z4 = {0, 0, 0, 0};
  int32x8 z8 = {0, 0, 0, 0, 0, 0, 0, 0};
  // 6-arg variant (clang-23 / therock-10.0 lane).
  __builtin_amdgcn_tensor_load_to_lds(g0, g1, z4, z4, z8, 0);
}
#endif

// ---------------------------------------------------------------------------
// GEMM: C = X @ W^T + bias.  128x128 workgroup tile, K=64 staging, 8 waves in
// 4x2 grid, each wave 32x64 (2x4 WMMA tiles) -> 16 WMMA per barrier pair.
// mode 0: bf16 head-major [B,H,L,DH] out.  mode 1: f32 row-major out.
// ---------------------------------------------------------------------------
#define GTM 128
#define GTN 128
#define GLD 72  // padded pitch: 144B, 16B-aligned, conflict-free

__global__ __launch_bounds__(256) void gemm_xwT_kernel(
    const float* __restrict__ X, const float* __restrict__ W,
    const float* __restrict__ bias, bf16_t* __restrict__ outB,
    float* __restrict__ outF, int mode) {
  __shared__ bf16_t sA[GTM][GLD];
  __shared__ bf16_t sB[GTN][GLD];

  const int tid  = threadIdx.x;
  const int lane = tid & 31;
  const int w    = tid >> 5;
  const int m0   = blockIdx.y * GTM;
  const int n0   = blockIdx.x * GTN;
  const int wm   = (w & 3) * 32;
  const int wn   = (w >> 2) * 64;

  v8f acc[2][4] = {};

  for (int k0 = 0; k0 < D_; k0 += 64) {
    // Stage A/B tiles 128x64 f32 -> bf16 (coalesced float4, 8 per thread).
#pragma unroll
    for (int i = 0; i < 8; i++) {
      int idx = i * 256 + tid;
      int r = idx >> 4, c = (idx & 15) * 4;
      float4 f = *reinterpret_cast<const float4*>(
          X + (size_t)(m0 + r) * D_ + k0 + c);
      sA[r][c + 0] = (bf16_t)f.x; sA[r][c + 1] = (bf16_t)f.y;
      sA[r][c + 2] = (bf16_t)f.z; sA[r][c + 3] = (bf16_t)f.w;
    }
#pragma unroll
    for (int i = 0; i < 8; i++) {
      int idx = i * 256 + tid;
      int r = idx >> 4, c = (idx & 15) * 4;
      float4 f = *reinterpret_cast<const float4*>(
          W + (size_t)(n0 + r) * D_ + k0 + c);
      sB[r][c + 0] = (bf16_t)f.x; sB[r][c + 1] = (bf16_t)f.y;
      sB[r][c + 2] = (bf16_t)f.z; sB[r][c + 3] = (bf16_t)f.w;
    }
    __syncthreads();

#pragma unroll
    for (int kk = 0; kk < 2; kk++) {
      bf16x16 a0 = frag_ld(&sA[0][0], GLD, wm,      kk * 32, lane);
      bf16x16 a1 = frag_ld(&sA[0][0], GLD, wm + 16, kk * 32, lane);
#pragma unroll
      for (int j = 0; j < 4; j++) {
        bf16x16 bj = frag_ld(&sB[0][0], GLD, wn + j * 16, kk * 32, lane);
        acc[0][j] = wmma_bf16(a0, bj, acc[0][j]);
        acc[1][j] = wmma_bf16(a1, bj, acc[1][j]);
      }
    }
    __syncthreads();
  }

  // Epilogue. C/D layout: N = lane&15, M = v + 8*(lane>>4).
  const int nlo = lane & 15;
  const int mhi = (lane >> 4) << 3;
#pragma unroll
  for (int tm = 0; tm < 2; tm++) {
#pragma unroll
    for (int tn = 0; tn < 4; tn++) {
      int col  = n0 + wn + tn * 16 + nlo;
      float bc = bias[col];
#pragma unroll
      for (int v = 0; v < 8; v++) {
        int row   = m0 + wm + tm * 16 + mhi + v;
        float val = acc[tm][tn][v] + bc;
        if (mode == 0) {
          int bb = row >> 10, l = row & (L_ - 1);
          int hh = col >> 6,  d = col & (DH_ - 1);
          outB[(((size_t)(bb * H_ + hh)) * L_ + l) * DH_ + d] = (bf16_t)val;
        } else {
          outF[(size_t)row * D_ + col] = val;
        }
      }
    }
  }
}

// ---------------------------------------------------------------------------
// Fused attention: workgroup = one (b,h) x 32 query rows.
// Phase A: S = (Q K^T)/32; K chunks streamed by TDM, double-buffered so the
//          DMA overlaps WMMA (TENSORcnt / s_wait_tensorcnt).
// Phase B: mask + softmax (wave32 shuffles), single HBM write of attn.
// Phase C: O = P V; P straight from LDS (f32->bf16 frags), V transposed in LDS.
// ---------------------------------------------------------------------------
__global__ __launch_bounds__(256) void attn_fused_kernel(
    const bf16_t* __restrict__ Q, const bf16_t* __restrict__ K,
    const bf16_t* __restrict__ V, const int* __restrict__ mask,
    float* __restrict__ attnOut, float* __restrict__ ctx) {
  __shared__ float sS[32][L_];                    // 128 KB scores/probs
  __shared__ union {
    bf16_t k2[2][128][72];                        // K chunks, double buffered
    bf16_t vt[64][136];                           // V chunk, transposed
  } tile;                                         // 36 KB
  __shared__ bf16_t sQ[32][72];                   // 4.5 KB

  const int tid  = threadIdx.x;
  const int lane = tid & 31;
  const int w    = tid >> 5;
  const int m0   = blockIdx.x * 32;
  const int h    = blockIdx.y;
  const int b    = blockIdx.z;
  const int bh   = b * H_ + h;
  const bf16_t* Qp = Q + (size_t)bh * L_ * DH_;
  const bf16_t* Kp = K + (size_t)bh * L_ * DH_;
  const bf16_t* Vp = V + (size_t)bh * L_ * DH_;

  // Load Q block (32x64 bf16).
#pragma unroll
  for (int i = 0; i < 4; i++) {
    int idx = i * 256 + tid;
    int r = idx >> 5, c = (idx & 31) * 2;
    *reinterpret_cast<unsigned*>(&sQ[r][c]) =
        *reinterpret_cast<const unsigned*>(Qp + (size_t)(m0 + r) * DH_ + c);
  }

  // ---- Phase A: scores --------------------------------------------------
#if HAVE_TDM
  if (w == 0) {
    tdm_load_2d_bf16(lds_off(&tile.k2[0][0][0]), Kp, DH_, L_, DH_, 128, DH_);
    __builtin_amdgcn_s_wait_tensorcnt(0);
  }
  __syncthreads();
  for (int c = 0; c < 8; c++) {
    const int s0 = c * 128;
    const bf16_t* kb = &tile.k2[c & 1][0][0];
    if (w == 0 && c + 1 < 8) {  // prefetch next chunk into the other buffer
      tdm_load_2d_bf16(lds_off(&tile.k2[(c + 1) & 1][0][0]),
                       Kp + (size_t)(s0 + 128) * DH_, DH_, L_, DH_, 128, DH_);
    }
    v8f acc0 = {}, acc1 = {};
    const int cn = w * 16;
#pragma unroll
    for (int kk = 0; kk < 2; kk++) {
      bf16x16 bf = frag_ld(kb, 72, cn, kk * 32, lane);
      bf16x16 a0 = frag_ld(&sQ[0][0], 72, 0,  kk * 32, lane);
      bf16x16 a1 = frag_ld(&sQ[0][0], 72, 16, kk * 32, lane);
      acc0 = wmma_bf16(a0, bf, acc0);
      acc1 = wmma_bf16(a1, bf, acc1);
    }
    const int nlo = lane & 15, mhi = (lane >> 4) << 3;
#pragma unroll
    for (int v = 0; v < 8; v++) {
      sS[mhi + v][s0 + cn + nlo]      = acc0[v] * 0.03125f;  // 1/sqrt(1024)
      sS[16 + mhi + v][s0 + cn + nlo] = acc1[v] * 0.03125f;
    }
    if (w == 0) __builtin_amdgcn_s_wait_tensorcnt(0);
    __syncthreads();
  }
#else
  __syncthreads();
  for (int c = 0; c < 8; c++) {
    const int s0 = c * 128;
    bf16_t* kb = &tile.k2[0][0][0];
#pragma unroll
    for (int i = 0; i < 16; i++) {
      int idx = i * 256 + tid;
      int r = idx >> 5, cc = (idx & 31) * 2;
      *reinterpret_cast<unsigned*>(kb + r * 72 + cc) =
          *reinterpret_cast<const unsigned*>(Kp + (size_t)(s0 + r) * DH_ + cc);
    }
    __syncthreads();
    v8f acc0 = {}, acc1 = {};
    const int cn = w * 16;
#pragma unroll
    for (int kk = 0; kk < 2; kk++) {
      bf16x16 bf = frag_ld(kb, 72, cn, kk * 32, lane);
      bf16x16 a0 = frag_ld(&sQ[0][0], 72, 0,  kk * 32, lane);
      bf16x16 a1 = frag_ld(&sQ[0][0], 72, 16, kk * 32, lane);
      acc0 = wmma_bf16(a0, bf, acc0);
      acc1 = wmma_bf16(a1, bf, acc1);
    }
    const int nlo = lane & 15, mhi = (lane >> 4) << 3;
#pragma unroll
    for (int v = 0; v < 8; v++) {
      sS[mhi + v][s0 + cn + nlo]      = acc0[v] * 0.03125f;
      sS[16 + mhi + v][s0 + cn + nlo] = acc1[v] * 0.03125f;
    }
    __syncthreads();
  }
#endif

  // ---- Phase B: mask + softmax + write attn -----------------------------
#pragma unroll 1
  for (int rr = 0; rr < 4; rr++) {
    const int m = w * 4 + rr;
    const int* mrow = mask + ((size_t)b * L_ + (m0 + m)) * L_;
    float* srow = sS[m];

    float mx = -3.4e38f;
    for (int s = lane; s < L_; s += 32) {
      float vv = (mrow[s] == 0) ? -1e9f : srow[s];
      srow[s] = vv;
      mx = fmaxf(mx, vv);
    }
#pragma unroll
    for (int off = 16; off; off >>= 1) mx = fmaxf(mx, __shfl_xor(mx, off));

    float sum = 0.f;
    for (int s = lane; s < L_; s += 32) {
      float e = __expf(srow[s] - mx);
      srow[s] = e;
      sum += e;
    }
#pragma unroll
    for (int off = 16; off; off >>= 1) sum += __shfl_xor(sum, off);
    const float inv = 1.0f / sum;

    float* arow = attnOut + ((size_t)bh * L_ + (m0 + m)) * L_;
    for (int s = lane; s < L_; s += 32) {
      float p = srow[s] * inv;
      srow[s] = p;   // keep probabilities in LDS for P@V
      arow[s] = p;   // single HBM write of attn
    }
  }
  __syncthreads();

  // ---- Phase C: O = P @ V ----------------------------------------------
  const int mt = (w & 1) * 16;
  const int dt = (w >> 1) * 16;
  v8f oacc = {};
  for (int s0 = 0; s0 < L_; s0 += 128) {
    __syncthreads();  // tile reuse fence
#pragma unroll
    for (int i = 0; i < 16; i++) {
      int idx = i * 256 + tid;
      int r = idx >> 5, c2 = (idx & 31) * 2;
      union { unsigned u; bf16_t x[2]; } cv;
      cv.u = *reinterpret_cast<const unsigned*>(Vp + (size_t)(s0 + r) * DH_ + c2);
      tile.vt[c2][r]     = cv.x[0];
      tile.vt[c2 + 1][r] = cv.x[1];
    }
    __syncthreads();
#pragma unroll
    for (int kk = 0; kk < 4; kk++) {
      bf16x16 a  = frag_ld_f32(&sS[0][0], L_, mt, s0 + kk * 32, lane);
      bf16x16 bf = frag_ld(&tile.vt[0][0], 136, dt, kk * 32, lane);
      oacc = wmma_bf16(a, bf, oacc);
    }
  }
  {
    const int nlo = lane & 15, mhi = (lane >> 4) << 3;
#pragma unroll
    for (int v = 0; v < 8; v++) {
      int row = m0 + mt + mhi + v;
      ctx[((size_t)b * L_ + row) * D_ + h * DH_ + dt + nlo] = oacc[v];
    }
  }
}

// ---------------------------------------------------------------------------
extern "C" void kernel_launch(void* const* d_in, const int* in_sizes, int n_in,
                              void* d_out, int out_size, void* d_ws,
                              size_t ws_size, hipStream_t stream) {
  const float* q    = (const float*)d_in[0];
  const float* k    = (const float*)d_in[1];
  const float* v    = (const float*)d_in[2];
  const int*   msk  = (const int*)d_in[3];
  const float* wq   = (const float*)d_in[4];
  const float* bq   = (const float*)d_in[5];
  const float* wk   = (const float*)d_in[6];
  const float* bk   = (const float*)d_in[7];
  const float* wv   = (const float*)d_in[8];
  const float* bv   = (const float*)d_in[9];
  const float* wo   = (const float*)d_in[10];
  const float* bo   = (const float*)d_in[11];

  float* out     = (float*)d_out;
  float* attnOut = out + (size_t)B_ * L_ * D_;  // tuple: (out, attn) concat

  char* ws = (char*)d_ws;                              // 80 MB used:
  bf16_t* Qbf = (bf16_t*)(ws);                         // 16 MB
  bf16_t* Kbf = (bf16_t*)(ws + (size_t)16 * 1024 * 1024);
  bf16_t* Vbf = (bf16_t*)(ws + (size_t)32 * 1024 * 1024);
  float*  ctx = (float*)(ws + (size_t)48 * 1024 * 1024);  // 32 MB

  dim3 blk(256);
  dim3 gG(D_ / GTN, (B_ * L_) / GTM);  // (8, 64)
  gemm_xwT_kernel<<<gG, blk, 0, stream>>>(q, wq, bq, Qbf, nullptr, 0);
  gemm_xwT_kernel<<<gG, blk, 0, stream>>>(k, wk, bk, Kbf, nullptr, 0);
  gemm_xwT_kernel<<<gG, blk, 0, stream>>>(v, wv, bv, Vbf, nullptr, 0);

  dim3 gA(L_ / 32, H_, B_);  // (32, 16, 8)
  attn_fused_kernel<<<gA, blk, 0, stream>>>(Qbf, Kbf, Vbf, msk, attnOut, ctx);

  gemm_xwT_kernel<<<gG, blk, 0, stream>>>(ctx, wo, bo, nullptr, out, 1);
}
